// GlobalAttention_9706626089895
// MI455X (gfx1250) — matrix operand order repaired and checked by reading, not compiled, not run on hardware
//
#include <hip/hip_runtime.h>
#include <hip/hip_bf16.h>

// ---------------------------------------------------------------------------
// GlobalAttention (AlphaFold-style) for gfx1250 / MI455X.
// bf16 WMMA GEMMs + TDM (tensor_load_to_lds) staging of the X tile.
// ---------------------------------------------------------------------------

#define R_DIM 128
#define S_DIM 2048
#define C_DIM 256
#define H_DIM 8
#define D_DIM 32
#define NORM_Q 0.17677669529663687f   // 32^-0.5

typedef __attribute__((ext_vector_type(16))) __bf16 v16bf;
typedef __attribute__((ext_vector_type(8)))  float  v8f;
typedef __attribute__((ext_vector_type(4)))  unsigned int ui32x4;
typedef __attribute__((ext_vector_type(8)))  int i32x8;
typedef __attribute__((ext_vector_type(4)))  int i32x4;

#define HAVE_TDM __has_builtin(__builtin_amdgcn_tensor_load_to_lds)

union FragU {
    v16bf v;
    unsigned short us[16];
    unsigned long long d[4];
    uint4 q[2];
};

__device__ __forceinline__ unsigned short f2bf(float f) {
    unsigned int u = __float_as_uint(f);
    unsigned int r = u + 0x7FFFu + ((u >> 16) & 1u);   // round-to-nearest-even
    return (unsigned short)(r >> 16);
}

// A-matrix fragment (16x32 bf16) from a row-major bf16 LDS tile.
// ISA 16-bit A layout: lane half hi gets K chunks {8hi..8hi+7, 16+8hi..16+8hi+7}.
__device__ __forceinline__ v16bf load_a_frag(const unsigned short* row, int kb, int hi) {
    FragU f;
    const unsigned short* p = row + kb * 32 + hi * 8;
    f.d[0] = *(const unsigned long long*)(p);
    f.d[1] = *(const unsigned long long*)(p + 4);
    f.d[2] = *(const unsigned long long*)(p + 16);
    f.d[3] = *(const unsigned long long*)(p + 20);
    return f.v;
}

// B-matrix fragment (32x16 bf16) from pre-packed fragment buffer:
// layout [ntile][kblock][lane][16 bf16] -> 32 contiguous bytes per lane.
__device__ __forceinline__ v16bf load_b_frag(const unsigned short* base, int nt, int kb, int lane) {
    FragU f;
    const uint4* p = (const uint4*)(base + (size_t)(((nt * 8) + kb) * 32 + lane) * 16);
    f.q[0] = p[0];
    f.q[1] = p[1];
    return f.v;
}

__device__ __forceinline__ v8f wmma_bf16(v16bf a, v16bf b, v8f c) {
    return __builtin_amdgcn_wmma_f32_16x16x32_bf16(false, a, false, b, (short)0, c, false, false);
}

// ---------------------------------------------------------------------------
// K0: repack weights into WMMA B-fragment layout (bf16).
// B element for (nt, kb, lane, j):  n = nt*16 + lane%16,  k = kb*32 + (j<8 ? 8*(lane/16) + j : 16 + 8*(lane/16) + j-8)
// mode 0: src row-major [K][srcN]; mode 1: concat wk|wv, srcN = 32 each.
// ---------------------------------------------------------------------------
__global__ void prep_frag_kernel(const float* __restrict__ src, const float* __restrict__ src2,
                                 unsigned short* __restrict__ dst, int NT, int srcN, int mode) {
    int idx = blockIdx.x * blockDim.x + threadIdx.x;
    int total = NT * 8 * 32 * 16;
    if (idx >= total) return;
    int j    = idx & 15;
    int lane = (idx >> 4) & 31;
    int kb   = (idx >> 9) & 7;
    int nt   = idx >> 12;
    int n    = nt * 16 + (lane & 15);
    int k    = kb * 32 + ((j < 8) ? ((lane >> 4) * 8 + j) : (16 + (lane >> 4) * 8 + (j - 8)));
    float val;
    if (mode == 0) {
        val = src[(size_t)k * srcN + n];
    } else {
        val = (n < 32) ? src[(size_t)k * 32 + n] : src2[(size_t)k * 32 + (n - 32)];
    }
    dst[idx] = f2bf(val);
}

// ---------------------------------------------------------------------------
// K1: masked mean pool over S:  q_pool[r][c]
// ---------------------------------------------------------------------------
__global__ void pool_kernel(const float* __restrict__ x, const float* __restrict__ mask,
                            float* __restrict__ qpool) {
    int r = blockIdx.x;
    int t = threadIdx.x;
    __shared__ float mlds[S_DIM];
    __shared__ float red[256];
    float mpart = 0.0f;
    for (int i = 0; i < S_DIM / 256; ++i) {
        float m = mask[(size_t)r * S_DIM + t + 256 * i];
        mlds[t + 256 * i] = m;
        mpart += m;
    }
    red[t] = mpart;
    __syncthreads();
    for (int s = 128; s > 0; s >>= 1) {
        if (t < s) red[t] += red[t + s];
        __syncthreads();
    }
    float inv = 1.0f / (red[0] + 1e-10f);
    float acc = 0.0f;
    const float* xr = x + (size_t)r * S_DIM * C_DIM + t;
    for (int s = 0; s < S_DIM; ++s)
        acc += xr[(size_t)s * C_DIM] * mlds[s];
    qpool[(size_t)r * C_DIM + t] = acc * inv;
}

// ---------------------------------------------------------------------------
// K2: q = (q_pool @ wq) * NORM   [128 x 256]
// ---------------------------------------------------------------------------
__global__ void qproj_kernel(const float* __restrict__ qpool, const float* __restrict__ wq,
                             float* __restrict__ q) {
    int r = blockIdx.x;
    int n = threadIdx.x;
    __shared__ float qp[C_DIM];
    qp[n] = qpool[(size_t)r * C_DIM + n];
    __syncthreads();
    float acc = 0.0f;
    for (int c = 0; c < C_DIM; ++c)
        acc += qp[c] * wq[(size_t)c * C_DIM + n];
    q[(size_t)r * C_DIM + n] = acc * NORM_Q;
}

// ---------------------------------------------------------------------------
// K3: k = x@wk, v = x@wv via WMMA.  64 tokens/block, 8 waves.
// ---------------------------------------------------------------------------
__global__ void __launch_bounds__(256) kv_kernel(const float* __restrict__ x,
                                                 const unsigned short* __restrict__ kvF,
                                                 float* __restrict__ kbuf, float* __restrict__ vbuf) {
    __shared__ unsigned short Xt[64 * C_DIM];
    int blk = blockIdx.x;
    int tid = threadIdx.x;
    size_t tok0 = (size_t)blk * 64;
    int r  = blk >> 5;          // 2048/64 = 32 blocks per r
    int s0 = (blk & 31) * 64;

    const float4* xg = (const float4*)(x + tok0 * C_DIM);
    #pragma unroll
    for (int i = 0; i < 16; ++i) {
        if (i < 15) __builtin_prefetch(xg + tid + 256 * (i + 1), 0, 1);
        float4 f = xg[tid + 256 * i];
        int base = (tid + 256 * i) * 4;
        Xt[base + 0] = f2bf(f.x);
        Xt[base + 1] = f2bf(f.y);
        Xt[base + 2] = f2bf(f.z);
        Xt[base + 3] = f2bf(f.w);
    }
    __syncthreads();

    int w = tid >> 5, lane = tid & 31;
    int strip = w >> 1, ng = w & 1;
    int hi = lane >> 4, ln = lane & 15;
    const unsigned short* arow = Xt + (strip * 16 + ln) * C_DIM;

    v16bf afrag[8];
    #pragma unroll
    for (int kb = 0; kb < 8; ++kb) afrag[kb] = load_a_frag(arow, kb, hi);

    #pragma unroll
    for (int nti = 0; nti < 2; ++nti) {
        int nt = ng * 2 + nti;
        v8f acc = {0.f, 0.f, 0.f, 0.f, 0.f, 0.f, 0.f, 0.f};
        #pragma unroll
        for (int kb = 0; kb < 8; ++kb) {
            v16bf b = load_b_frag(kvF, nt, kb, lane);
            acc = wmma_bf16(afrag[kb], b, acc);
        }
        int n = nt * 16 + ln;
        #pragma unroll
        for (int p = 0; p < 8; ++p) {
            int srow = s0 + strip * 16 + hi * 8 + p;
            float val = acc[p];
            if (n < D_DIM)
                kbuf[((size_t)r * S_DIM + srow) * D_DIM + n] = val;
            else
                vbuf[((size_t)r * S_DIM + srow) * D_DIM + (n - D_DIM)] = val;
        }
    }
}

// ---------------------------------------------------------------------------
// K4: attention per r: logits -> softmax -> o = attn @ v.
// ---------------------------------------------------------------------------
__global__ void __launch_bounds__(256) attn_kernel(const float* __restrict__ qws,
                                                   const float* __restrict__ mask,
                                                   const float* __restrict__ kbuf,
                                                   const float* __restrict__ vbuf,
                                                   float* __restrict__ ows) {
    int r = blockIdx.x;
    int t = threadIdx.x;
    __shared__ float qs[C_DIM];
    __shared__ float logits[H_DIM][S_DIM];

    qs[t] = qws[(size_t)r * C_DIM + t];
    __syncthreads();

    // Phase 1: logits[h][s] with mask bias.
    for (int i = 0; i < S_DIM / 256; ++i) {
        int s = t + 256 * i;
        const float4* kp = (const float4*)(kbuf + ((size_t)r * S_DIM + s) * D_DIM);
        float4 kq[8];
        #pragma unroll
        for (int u = 0; u < 8; ++u) kq[u] = kp[u];
        float m = mask[(size_t)r * S_DIM + s];
        float bias = (m == 0.0f) ? -1.0e9f : 0.0f;
        #pragma unroll
        for (int h = 0; h < H_DIM; ++h) {
            float acc = 0.0f;
            #pragma unroll
            for (int u = 0; u < 8; ++u) {
                float4 kk = kq[u];
                acc += qs[h * 32 + u * 4 + 0] * kk.x;
                acc += qs[h * 32 + u * 4 + 1] * kk.y;
                acc += qs[h * 32 + u * 4 + 2] * kk.z;
                acc += qs[h * 32 + u * 4 + 3] * kk.w;
            }
            logits[h][s] = acc + bias;
        }
    }
    __syncthreads();

    // Phase 2: softmax, one wave32 per head.
    int h = t >> 5, l = t & 31;
    float mx = -3.0e38f;
    for (int s = l; s < S_DIM; s += 32) mx = fmaxf(mx, logits[h][s]);
    #pragma unroll
    for (int off = 16; off > 0; off >>= 1) mx = fmaxf(mx, __shfl_xor(mx, off, 32));
    float sum = 0.0f;
    for (int s = l; s < S_DIM; s += 32) {
        float e = __expf(logits[h][s] - mx);
        logits[h][s] = e;
        sum += e;
    }
    #pragma unroll
    for (int off = 16; off > 0; off >>= 1) sum += __shfl_xor(sum, off, 32);
    float inv = 1.0f / sum;
    for (int s = l; s < S_DIM; s += 32) logits[h][s] *= inv;
    __syncthreads();

    // Phase 3: o[h][d] = sum_s attn[h][s] * v[s][d];  (h = t>>5, d = l)
    float acc = 0.0f;
    for (int s = 0; s < S_DIM; ++s)
        acc += logits[h][s] * vbuf[((size_t)r * S_DIM + s) * D_DIM + l];
    ows[(size_t)r * C_DIM + h * 32 + l] = acc;
}

// ---------------------------------------------------------------------------
// K5: out = (sigmoid(x@wg + bg) * o_gate) @ wo + bo   -- two fused WMMA GEMMs.
// X tile (64 tokens x 256 ch fp32, 64 KB contiguous) staged into LDS by the
// Tensor Data Mover (one wave issues tensor_load_to_lds, waits TENSORcnt),
// then converted fp32->bf16 in LDS.  The fp32 staging buffer is reused for
// the gated intermediate T (bf16) after conversion.
// ---------------------------------------------------------------------------
__global__ void __launch_bounds__(256) gate_out_kernel(const float* __restrict__ x,
                                                       const float* __restrict__ bg,
                                                       const float* __restrict__ bo,
                                                       const float* __restrict__ ows,
                                                       const unsigned short* __restrict__ wgF,
                                                       const unsigned short* __restrict__ woF,
                                                       float* __restrict__ out) {
    // LDS layout:
    //   [0,      65536)  : Xf   fp32 staging (dead after conversion)  / Tt bf16 reuse
    //   [65536,  98304)  : Xt   bf16 X tile
    //   [98304, 101376)  : og | bgs | bos  (f32, 256 each)
    __shared__ __align__(16) char smem[101376];
    float*          Xf = (float*)smem;
    unsigned short* Xt = (unsigned short*)(smem + 65536);
    unsigned short* Tt = (unsigned short*)smem;            // aliases Xf (dead)
    float*          og = (float*)(smem + 98304);
    float*          bgs = og + C_DIM;
    float*          bos = bgs + C_DIM;

    int blk = blockIdx.x;
    int tid = threadIdx.x;
    size_t tok0 = (size_t)blk * 64;
    int r = blk >> 5;

#if HAVE_TDM
    // ---- TDM staging: one wave issues the descriptor; EXEC is ignored. ----
    if (tid < 32) {
        unsigned long long gaddr = (unsigned long long)(const void*)(x + tok0 * C_DIM);
        unsigned lds_off = (unsigned)(unsigned long long)(void*)Xf;  // low 32b = LDS byte offset
        // D# group 0: count=1 | lds_addr | global_addr(57b) | type=2
        ui32x4 g0;
        g0[0] = 1u;
        g0[1] = lds_off;
        g0[2] = (unsigned)(gaddr & 0xFFFFFFFFull);
        g0[3] = (unsigned)((gaddr >> 32) & 0x01FFFFFFull) | (2u << 30);
        // D# group 1: data_size=4B (code 2); 1-D tile: tensor_dim0=tile_dim0=16384,
        // tensor_dim1=tile_dim1=1, stride0=16384.
        i32x8 g1;
        g1[0] = (int)(2u << 16);          // wg_mask=0, data_size=2 (4 bytes)
        g1[1] = (int)(16384u << 16);      // [63:48] = tensor_dim0[15:0]
        g1[2] = (int)(1u << 16);          // [79:64]=td0 hi=0, [95:80]=tensor_dim1 lo=1
        g1[3] = (int)(16384u << 16);      // [111:96]=td1 hi=0, [127:112]=tile_dim0
        g1[4] = 1;                        // tile_dim1=1, tile_dim2=0
        g1[5] = 16384;                    // tensor_dim0_stride lo32
        g1[6] = 0;
        g1[7] = 0;
        i32x4 gz4 = {0, 0, 0, 0};
        i32x8 gz8 = {0, 0, 0, 0, 0, 0, 0, 0};
        // 6-arg variant (clang-23 / therock-10.0 headers): trailing group unused here.
        __builtin_amdgcn_tensor_load_to_lds(g0, g1, gz4, gz4, gz8, 0);
#if __has_builtin(__builtin_amdgcn_s_wait_tensorcnt)
        __builtin_amdgcn_s_wait_tensorcnt(0);
#else
        asm volatile("s_wait_tensorcnt 0x0" ::: "memory");
#endif
    }
    og[tid]  = ows[(size_t)r * C_DIM + tid];
    bgs[tid] = bg[tid];
    bos[tid] = bo[tid];
    __syncthreads();
    // Convert fp32 LDS tile -> bf16 LDS tile.
    #pragma unroll
    for (int i = 0; i < 64; ++i) {
        int idx = tid + 256 * i;
        Xt[idx] = f2bf(Xf[idx]);
    }
    __syncthreads();
#else
    // ---- Fallback: vector-load staging with in-register conversion. ----
    const float4* xg = (const float4*)(x + tok0 * C_DIM);
    #pragma unroll
    for (int i = 0; i < 16; ++i) {
        float4 f = xg[tid + 256 * i];
        int base = (tid + 256 * i) * 4;
        Xt[base + 0] = f2bf(f.x);
        Xt[base + 1] = f2bf(f.y);
        Xt[base + 2] = f2bf(f.z);
        Xt[base + 3] = f2bf(f.w);
    }
    og[tid]  = ows[(size_t)r * C_DIM + tid];
    bgs[tid] = bg[tid];
    bos[tid] = bo[tid];
    __syncthreads();
#endif

    int w = tid >> 5, lane = tid & 31;
    int strip = w >> 1, ng = w & 1;
    int hi = lane >> 4, ln = lane & 15;

    // GEMM1: G = X @ Wg  -> sigmoid gate -> T (bf16 in LDS, aliasing Xf)
    {
        const unsigned short* arow = Xt + (strip * 16 + ln) * C_DIM;
        v16bf afrag[8];
        #pragma unroll
        for (int kb = 0; kb < 8; ++kb) afrag[kb] = load_a_frag(arow, kb, hi);

        #pragma unroll
        for (int nti = 0; nti < 8; ++nti) {
            int nt = ng * 8 + nti;
            v8f acc = {0.f, 0.f, 0.f, 0.f, 0.f, 0.f, 0.f, 0.f};
            #pragma unroll
            for (int kb = 0; kb < 8; ++kb) {
                v16bf b = load_b_frag(wgF, nt, kb, lane);
                acc = wmma_bf16(afrag[kb], b, acc);
            }
            int n = nt * 16 + ln;
            float gate_o = og[n];
            float bias   = bgs[n];
            #pragma unroll
            for (int p = 0; p < 8; ++p) {
                int trow = strip * 16 + hi * 8 + p;
                float pre = acc[p] + bias;
                float g = 1.0f / (1.0f + __expf(-pre));
                Tt[trow * C_DIM + n] = f2bf(g * gate_o);
            }
        }
    }
    __syncthreads();

    // GEMM2: Out = T @ Wo + bo
    {
        const unsigned short* trow_p = Tt + (strip * 16 + ln) * C_DIM;
        v16bf tfrag[8];
        #pragma unroll
        for (int kb = 0; kb < 8; ++kb) tfrag[kb] = load_a_frag(trow_p, kb, hi);

        #pragma unroll
        for (int nti = 0; nti < 8; ++nti) {
            int nt = ng * 8 + nti;
            v8f acc = {0.f, 0.f, 0.f, 0.f, 0.f, 0.f, 0.f, 0.f};
            #pragma unroll
            for (int kb = 0; kb < 8; ++kb) {
                v16bf b = load_b_frag(woF, nt, kb, lane);
                acc = wmma_bf16(tfrag[kb], b, acc);
            }
            int n = nt * 16 + ln;
            float bias = bos[n];
            #pragma unroll
            for (int p = 0; p < 8; ++p) {
                int trow = strip * 16 + hi * 8 + p;
                out[(tok0 + trow) * C_DIM + n] = acc[p] + bias;
            }
        }
    }
}

// ---------------------------------------------------------------------------
// Launcher
// ---------------------------------------------------------------------------
extern "C" void kernel_launch(void* const* d_in, const int* in_sizes, int n_in,
                              void* d_out, int out_size, void* d_ws, size_t ws_size,
                              hipStream_t stream) {
    const float* x    = (const float*)d_in[0];
    const float* mask = (const float*)d_in[1];
    const float* wq   = (const float*)d_in[2];
    const float* wk   = (const float*)d_in[3];
    const float* wv   = (const float*)d_in[4];
    const float* wg   = (const float*)d_in[5];
    const float* bg   = (const float*)d_in[6];
    const float* wo   = (const float*)d_in[7];
    const float* bo   = (const float*)d_in[8];
    float* out = (float*)d_out;

    char* ws = (char*)d_ws;
    // Workspace layout (bytes):
    const size_t off_qpool = 0;                                  // 131072
    const size_t off_q     = off_qpool + 131072;                 // 131072
    const size_t off_k     = off_q + 131072;                     // 33554432
    const size_t off_v     = off_k + 33554432;                   // 33554432
    const size_t off_o     = off_v + 33554432;                   // 131072
    const size_t off_wgF   = off_o + 131072;                     // 131072
    const size_t off_woF   = off_wgF + 131072;                   // 131072
    const size_t off_kvF   = off_woF + 131072;                   // 32768

    float* qpool = (float*)(ws + off_qpool);
    float* q     = (float*)(ws + off_q);
    float* kbuf  = (float*)(ws + off_k);
    float* vbuf  = (float*)(ws + off_v);
    float* ows   = (float*)(ws + off_o);
    unsigned short* wgF = (unsigned short*)(ws + off_wgF);
    unsigned short* woF = (unsigned short*)(ws + off_woF);
    unsigned short* kvF = (unsigned short*)(ws + off_kvF);

    // K0: pack weights into WMMA B-fragment layout.
    prep_frag_kernel<<<256, 256, 0, stream>>>(wg, nullptr, wgF, 16, 256, 0);
    prep_frag_kernel<<<256, 256, 0, stream>>>(wo, nullptr, woF, 16, 256, 0);
    prep_frag_kernel<<<64, 256, 0, stream>>>(wk, wv, kvF, 4, 32, 1);

    // K1/K2: pooled query projection.
    pool_kernel<<<R_DIM, 256, 0, stream>>>(x, mask, qpool);
    qproj_kernel<<<R_DIM, 256, 0, stream>>>(qpool, wq, q);

    // K3: k/v projection (WMMA).
    kv_kernel<<<(R_DIM * S_DIM) / 64, 256, 0, stream>>>(x, kvF, kbuf, vbuf);

    // K4: attention + softmax + o.
    attn_kernel<<<R_DIM, 256, 0, stream>>>(q, mask, kbuf, vbuf, ows);

    // K5: fused gate GEMM + output GEMM (WMMA + TDM staging).
    gate_out_kernel<<<(R_DIM * S_DIM) / 64, 256, 0, stream>>>(x, bg, bo, ows, wgF, woF, out);
}
